// LawinTransformerPixelDecoder_4269197492626
// MI455X (gfx1250) — matrix operand order repaired and checked by reading, not compiled, not run on hardware
//
#include <hip/hip_runtime.h>

typedef __attribute__((ext_vector_type(16))) __bf16 v16bf;
typedef __attribute__((ext_vector_type(8)))  float  v8f;

#define CDIV(a,b) (((a)+(b)-1)/(b))

__device__ __forceinline__ __bf16 f2bf(float x) {
  unsigned u = __builtin_bit_cast(unsigned, x);
  unsigned short s = (unsigned short)((u + 0x7FFFu + ((u >> 16) & 1u)) >> 16); // RNE
  return __builtin_bit_cast(__bf16, s);
}

// pack two f32 -> two bf16 in one dword (HW pack-convert when available)
__device__ __forceinline__ unsigned pk2(float a, float b) {
#if defined(__gfx1250__) && __has_builtin(__builtin_amdgcn_cvt_pk_bf16_f32)
  auto p = __builtin_amdgcn_cvt_pk_bf16_f32(a, b);   // v_cvt_pk_bf16_f32
  return __builtin_bit_cast(unsigned, p);
#else
  unsigned ua = __builtin_bit_cast(unsigned, a);
  unsigned ub = __builtin_bit_cast(unsigned, b);
  unsigned ra = (ua + 0x7FFFu + ((ua >> 16) & 1u)) >> 16;
  unsigned rb = (ub + 0x7FFFu + ((ub >> 16) & 1u)) >> 16;
  return (ra & 0xFFFFu) | (rb << 16);
#endif
}

// ---------------------------------------------------------------------------
// TDM probe: NULL tensor descriptor (count=0 => architectural no-op) just to
// exercise tensor_load_to_lds + s_wait_tensorcnt. Guarded so it vanishes on
// toolchains without the builtin (and on the host pass).
// ---------------------------------------------------------------------------
__global__ void tdm_probe() {
#if defined(__gfx1250__) && __has_builtin(__builtin_amdgcn_tensor_load_to_lds)
  typedef unsigned int u32x4 __attribute__((ext_vector_type(4)));
  typedef int i32x4 __attribute__((ext_vector_type(4)));
  typedef int i32x8 __attribute__((ext_vector_type(8)));
  u32x4 g0 = {0u, 0u, 0u, 0u};          // count=0 -> NULL tensor
  i32x8 g1 = {0, 0, 0, 0, 0, 0, 0, 0};
  i32x4 g2 = {0, 0, 0, 0};
  i32x4 g3 = {0, 0, 0, 0};
#if __clang_major__ >= 23
  i32x8 g4 = {0, 0, 0, 0, 0, 0, 0, 0};
  __builtin_amdgcn_tensor_load_to_lds(g0, g1, g2, g3, g4, 0);
#else
  __builtin_amdgcn_tensor_load_to_lds(g0, g1, g2, g3, 0);
#endif
#if __has_builtin(__builtin_amdgcn_s_wait_tensorcnt)
  __builtin_amdgcn_s_wait_tensorcnt(0);
#endif
#endif
}

// ---------------------------------------------------------------------------
// Generic bf16-in / f32-accumulate WMMA GEMM, double-buffered LDS.
//   C[m, n] = act( A[m,:] * B[:, n] + bias[n] (+ C if accum) )
// A: f32 row-major [M x K], row stride lda (multiple of 4)
// Bw: transB==0 -> [K x N] row stride ldb ; transB==1 -> [N x K] (A @ Bw^T)
// Block tile 128x128, 8 waves of 32x64 (2x4 WMMA tiles), K-step 32.
// ---------------------------------------------------------------------------
#define TM 128
#define TN 128
#define TK 32

__global__ __launch_bounds__(256) void gemm_wmma(
    const float* __restrict__ A, int lda,
    const float* __restrict__ Bw, int ldb,
    const float* __restrict__ bias,
    float* __restrict__ C, long long ldc, long long colOff,
    int M, int N, int K,
    int transB, int doRelu, int accum, int outNCHW, int HW)
{
  __shared__ alignas(16) __bf16 As[2][TM][TK + 4];   // 36 * 2B rows (8B aligned)
  __shared__ alignas(16) __bf16 Bs[2][TK][TN + 4];   // 132 * 2B rows (8B aligned)

  const int tid  = threadIdx.x;
  const int lane = tid & 31;
  const int wave = tid >> 5;
  const int m0 = blockIdx.y * TM;
  const int n0 = blockIdx.x * TN;
  const int wm = (wave >> 1) * 32;   // 4 waves along M
  const int wn = (wave & 1) * 64;    // 2 waves along N

  v8f zero = {};
  v8f acc[2][4];
#pragma unroll
  for (int i = 0; i < 2; ++i)
#pragma unroll
    for (int j = 0; j < 4; ++j) acc[i][j] = zero;

  const int mrow  = lane & 15;
  const int halfk = (lane >> 4) << 3;
  const int nlane = lane & 15;
  const int mhalf = (lane >> 4) << 3;

  float4 ra[4], rb[4];

  // ---- staging helpers (vector fast path, guarded slow path) ----
  auto loadA = [&](int k0) {
#pragma unroll
    for (int s = 0; s < 4; ++s) {
      int slot = tid + s * 256;            // 1024 slots: 128 rows x 8 float4
      int r = slot >> 3, c4 = (slot & 7) << 2;
      int gm = m0 + r, gk = k0 + c4;
      float4 v = {0.f, 0.f, 0.f, 0.f};
      if (gm < M) {
        if (gk + 3 < K) {
          v = *(const float4*)(A + (long long)gm * lda + gk);
        } else {
          float* pv = &v.x;
          for (int t = 0; t < 4; ++t)
            if (gk + t < K) pv[t] = A[(long long)gm * lda + gk + t];
        }
      }
      ra[s] = v;
    }
  };
  auto loadB = [&](int k0) {
#pragma unroll
    for (int s = 0; s < 4; ++s) {
      int slot = tid + s * 256;
      float4 v = {0.f, 0.f, 0.f, 0.f};
      if (!transB) {                       // 32 rows x 32 float4 along N
        int r = slot >> 5, c4 = (slot & 31) << 2;
        int gk = k0 + r, gn = n0 + c4;
        if (gk < K) {
          if (gn + 3 < N) {
            v = *(const float4*)(Bw + (long long)gk * ldb + gn);
          } else {
            float* pv = &v.x;
            for (int t = 0; t < 4; ++t)
              if (gn + t < N) pv[t] = Bw[(long long)gk * ldb + gn + t];
          }
        }
      } else {                             // 128 n-rows x 8 float4 along K
        int n = slot >> 3, kq = (slot & 7) << 2;
        int gn = n0 + n, gk = k0 + kq;
        if (gn < N) {
          if (gk + 3 < K) {
            v = *(const float4*)(Bw + (long long)gn * ldb + gk);
          } else {
            float* pv = &v.x;
            for (int t = 0; t < 4; ++t)
              if (gk + t < K) pv[t] = Bw[(long long)gn * ldb + gk + t];
          }
        }
      }
      rb[s] = v;
    }
  };
  auto storeA = [&](int buf) {
#pragma unroll
    for (int s = 0; s < 4; ++s) {
      int slot = tid + s * 256;
      int r = slot >> 3, c4 = (slot & 7) << 2;
      uint2 u; u.x = pk2(ra[s].x, ra[s].y); u.y = pk2(ra[s].z, ra[s].w);
      *(uint2*)&As[buf][r][c4] = u;        // ds_store_b64
    }
  };
  auto storeB = [&](int buf) {
#pragma unroll
    for (int s = 0; s < 4; ++s) {
      int slot = tid + s * 256;
      if (!transB) {
        int r = slot >> 5, c4 = (slot & 31) << 2;
        uint2 u; u.x = pk2(rb[s].x, rb[s].y); u.y = pk2(rb[s].z, rb[s].w);
        *(uint2*)&Bs[buf][r][c4] = u;
      } else {                             // transpose-in-LDS: 4 b16 stores
        int n = slot >> 3, kq = (slot & 7) << 2;
        Bs[buf][kq + 0][n] = f2bf(rb[s].x);
        Bs[buf][kq + 1][n] = f2bf(rb[s].y);
        Bs[buf][kq + 2][n] = f2bf(rb[s].z);
        Bs[buf][kq + 3][n] = f2bf(rb[s].w);
      }
    }
  };

  const int nk = CDIV(K, TK);
  loadA(0); loadB(0);
  storeA(0); storeB(0);
  __syncthreads();

  for (int kt = 0; kt < nk; ++kt) {
    const int cur = kt & 1, nxt = cur ^ 1;
    if (kt + 1 < nk) { loadA((kt + 1) * TK); loadB((kt + 1) * TK); }
    if (kt + 2 < nk) {                     // global_prefetch_b8 two tiles ahead
      int pm = m0 + (tid & 127);
      if (pm < M) __builtin_prefetch(&A[(long long)pm * lda + (kt + 2) * TK], 0, 1);
    }

    // fragments per documented gfx1250 16-bit WMMA layouts
    v16bf af[2], bfr[4];
#pragma unroll
    for (int t = 0; t < 2; ++t)
#pragma unroll
      for (int v = 0; v < 8; ++v) {
        int k = ((v & 4) << 2) + ((v & 3) << 1) + halfk;
        af[t][2 * v]     = As[cur][wm + t * 16 + mrow][k];
        af[t][2 * v + 1] = As[cur][wm + t * 16 + mrow][k + 1];
      }
#pragma unroll
    for (int t = 0; t < 4; ++t)
#pragma unroll
      for (int v = 0; v < 8; ++v) {
        int k = ((v & 4) << 2) + ((v & 3) << 1) + halfk;
        bfr[t][2 * v]     = Bs[cur][k][wn + t * 16 + nlane];
        bfr[t][2 * v + 1] = Bs[cur][k + 1][wn + t * 16 + nlane];
      }
#pragma unroll
    for (int i = 0; i < 2; ++i)
#pragma unroll
      for (int j = 0; j < 4; ++j)
        acc[i][j] = __builtin_amdgcn_wmma_f32_16x16x32_bf16(
            false, af[i], false, bfr[j], (short)0, acc[i][j], false, false);

    if (kt + 1 < nk) { storeA(nxt); storeB(nxt); }
    __syncthreads();
  }

  // epilogue
#pragma unroll
  for (int i = 0; i < 2; ++i)
#pragma unroll
    for (int j = 0; j < 4; ++j)
#pragma unroll
      for (int rr = 0; rr < 8; ++rr) {
        int gm = m0 + wm + i * 16 + mhalf + rr;
        int gn = n0 + wn + j * 16 + nlane;
        if (gm < M && gn < N) {
          long long idx;
          if (outNCHW) {
            int b = gm / HW, hw = gm % HW;
            idx = ((long long)b * N + gn) * HW + hw;
          } else {
            idx = (long long)gm * ldc + colOff + gn;
          }
          float v = acc[i][j][rr];
          if (bias) v += bias[gn];
          if (accum) v += C[idx];
          if (doRelu) v = fmaxf(v, 0.f);
          C[idx] = v;
        }
      }
}

// ---------------------------------------------------------------------------
// Fused windowed attention, one block = one (window, head): S=64, d_head=64
// QP/KP/VP/O: [32768 x 512] f32, head h occupies columns [64h, 64h+64)
// ---------------------------------------------------------------------------
__global__ __launch_bounds__(256) void attn64(const float* __restrict__ QP,
                                              const float* __restrict__ KP,
                                              const float* __restrict__ VP,
                                              float* __restrict__ O)
{
  __shared__ alignas(16) __bf16 Qs[64][68];
  __shared__ alignas(16) __bf16 Ks[64][68];
  __shared__ alignas(16) __bf16 Vs[64][68];
  __shared__ float Ss[64][65];

  const int tid  = threadIdx.x;
  const int lane = tid & 31;
  const int wave = tid >> 5;
  const long long rowBase = (long long)(blockIdx.x >> 3) * 64;
  const int colBase = (blockIdx.x & 7) * 64;

  for (int s = tid; s < 1024; s += 256) {    // 64 rows x 16 float4
    int r = s >> 4, c4 = (s & 15) << 2;
    long long g = (rowBase + r) * 512 + colBase + c4;
    float4 q = *(const float4*)(QP + g);
    float4 k = *(const float4*)(KP + g);
    float4 v = *(const float4*)(VP + g);
    uint2 u;
    u.x = pk2(q.x, q.y); u.y = pk2(q.z, q.w); *(uint2*)&Qs[r][c4] = u;
    u.x = pk2(k.x, k.y); u.y = pk2(k.z, k.w); *(uint2*)&Ks[r][c4] = u;
    u.x = pk2(v.x, v.y); u.y = pk2(v.z, v.w); *(uint2*)&Vs[r][c4] = u;
  }
  __syncthreads();

  const int mrow  = lane & 15;
  const int halfk = (lane >> 4) << 3;
  const int nlane = lane & 15;
  const int mhalf = (lane >> 4) << 3;

  // S = Q K^T * (1/sqrt(64));  16 tiles of 16x16, 2 per wave
#pragma unroll
  for (int tt = 0; tt < 2; ++tt) {
    int t = wave * 2 + tt;
    int tm = (t >> 2) << 4, tn = (t & 3) << 4;
    v8f acc = {};
#pragma unroll
    for (int k0 = 0; k0 < 64; k0 += 32) {
      v16bf a, b;
#pragma unroll
      for (int v = 0; v < 8; ++v) {
        int k = k0 + ((v & 4) << 2) + ((v & 3) << 1) + halfk;
        a[2 * v]     = Qs[tm + mrow][k];
        a[2 * v + 1] = Qs[tm + mrow][k + 1];
      }
#pragma unroll
      for (int v = 0; v < 8; ++v) {        // B[k][n] = K[n][k]
        int k = k0 + ((v & 4) << 2) + ((v & 3) << 1) + halfk;
        b[2 * v]     = Ks[tn + nlane][k];
        b[2 * v + 1] = Ks[tn + nlane][k + 1];
      }
      acc = __builtin_amdgcn_wmma_f32_16x16x32_bf16(false, a, false, b,
                                                    (short)0, acc, false, false);
    }
#pragma unroll
    for (int rr = 0; rr < 8; ++rr)
      Ss[tm + mhalf + rr][tn + nlane] = acc[rr] * 0.125f;
  }
  __syncthreads();

  // row softmax; P (bf16) overwrites Qs
  if (tid < 64) {
    float mx = -3.4e38f;
    for (int c = 0; c < 64; ++c) mx = fmaxf(mx, Ss[tid][c]);
    float s = 0.f;
    for (int c = 0; c < 64; ++c) { float e = __expf(Ss[tid][c] - mx); Ss[tid][c] = e; s += e; }
    float inv = 1.f / s;
    for (int c = 0; c < 64; ++c) Qs[tid][c] = f2bf(Ss[tid][c] * inv);
  }
  __syncthreads();

  // O = P V
#pragma unroll
  for (int tt = 0; tt < 2; ++tt) {
    int t = wave * 2 + tt;
    int tm = (t >> 2) << 4, tn = (t & 3) << 4;
    v8f acc = {};
#pragma unroll
    for (int k0 = 0; k0 < 64; k0 += 32) {
      v16bf a, b;
#pragma unroll
      for (int v = 0; v < 8; ++v) {
        int k = k0 + ((v & 4) << 2) + ((v & 3) << 1) + halfk;
        a[2 * v]     = Qs[tm + mrow][k];
        a[2 * v + 1] = Qs[tm + mrow][k + 1];
      }
#pragma unroll
      for (int v = 0; v < 8; ++v) {        // B[k][n] = V[k][n]
        int k = k0 + ((v & 4) << 2) + ((v & 3) << 1) + halfk;
        b[2 * v]     = Vs[k][tn + nlane];
        b[2 * v + 1] = Vs[k + 1][tn + nlane];
      }
      acc = __builtin_amdgcn_wmma_f32_16x16x32_bf16(false, a, false, b,
                                                    (short)0, acc, false, false);
    }
#pragma unroll
    for (int rr = 0; rr < 8; ++rr)
      O[(rowBase + tm + mhalf + rr) * 512 + colBase + tn + nlane] = acc[rr];
  }
}

// --------------------------- data-movement kernels -------------------------

__global__ void nchw_to_nhwc(const float* __restrict__ S, float* __restrict__ D,
                             int B, int C, int HW)
{
  long long idx = (long long)blockIdx.x * 256 + threadIdx.x;
  long long tot = (long long)B * C * HW;
  if (idx >= tot) return;
  int c = (int)(idx % C);
  long long t = idx / C;
  int p = (int)(t % HW);
  int b = (int)(t / HW);
  D[idx] = S[((long long)b * C + c) * HW + p];
}

__global__ void expand_nearest(const float* __restrict__ S, float* __restrict__ D,
                               int srcHW, int scale, int colOff)
{
  long long idx = (long long)blockIdx.x * 256 + threadIdx.x;
  if (idx >= 32768LL * 512) return;
  int c = (int)(idx % 512);
  long long pix = idx / 512;
  int b = (int)(pix >> 12);
  int p = (int)(pix & 4095);
  int y = p >> 6, x = p & 63;
  int sy = y / scale, sx = x / scale;
  D[pix * 1536 + colOff + c] =
      S[((long long)b * srcHW * srcHW + sy * srcHW + sx) * 512 + c];
}

__global__ void mean_pix(const float* __restrict__ Cb, float* __restrict__ Mb)
{
  int idx = blockIdx.x * 256 + threadIdx.x;   // 8*512
  if (idx >= 8 * 512) return;
  int c = idx % 512, b = idx / 512;
  float s0 = 0.f, s1 = 0.f, s2 = 0.f, s3 = 0.f;
  for (int p = 0; p < 4096; p += 4) {
    s0 += Cb[((long long)b * 4096 + p + 0) * 512 + c];
    s1 += Cb[((long long)b * 4096 + p + 1) * 512 + c];
    s2 += Cb[((long long)b * 4096 + p + 2) * 512 + c];
    s3 += Cb[((long long)b * 4096 + p + 3) * 512 + c];
  }
  Mb[idx] = (s0 + s1 + s2 + s3) * (1.f / 4096.f);
}

__global__ void bcast_pool(const float* __restrict__ Pb, float* __restrict__ CAT)
{
  long long idx = (long long)blockIdx.x * 256 + threadIdx.x;
  if (idx >= 32768LL * 512) return;
  int c = (int)(idx % 512);
  long long pix = idx / 512;
  int b = (int)(pix >> 12);
  CAT[pix * 2560 + 512 + c] = Pb[b * 512 + c];
}

__device__ __forceinline__ long long win2pix(long long row)
{
  int t = (int)(row & 63);
  long long w = row >> 6;
  int j = (int)(w & 7);
  int i = (int)((w >> 3) & 7);
  int b = (int)(w >> 6);
  int py = t >> 3, px = t & 7;
  return (long long)b * 4096 + (i * 8 + py) * 64 + (j * 8 + px);
}

__global__ void q_gather(const float* __restrict__ Cb, float* __restrict__ Q)
{
  long long idx = (long long)blockIdx.x * 256 + threadIdx.x;
  if (idx >= 32768LL * 512) return;
  int c = (int)(idx % 512);
  long long row = idx / 512;
  Q[idx] = Cb[win2pix(row) * 512 + c];
}

__global__ void unwin_add(const float* __restrict__ O2, const float* __restrict__ Cb,
                          float* __restrict__ CAT, int colOff)
{
  long long idx = (long long)blockIdx.x * 256 + threadIdx.x;
  if (idx >= 32768LL * 512) return;
  int c = (int)(idx % 512);
  long long row = idx / 512;
  long long pix = win2pix(row);
  CAT[pix * 2560 + colOff + c] = O2[idx] + Cb[pix * 512 + c];
}

// downsample conv: X NHWC [32768 x 512] -> Y NHWC [32768 x cp], kernel r x r,
// pad (r/2-1, r/2), stride 1, weights OIHW (cp,512,r,r), relu
__global__ void conv_ds(const float* __restrict__ X, const float* __restrict__ Wc,
                        const float* __restrict__ bc, float* __restrict__ Y,
                        int r, int cp)
{
  long long idx = (long long)blockIdx.x * 256 + threadIdx.x;
  long long tot = 32768LL * cp;
  if (idx >= tot) return;
  int co = (int)(idx % cp);
  long long pix = idx / cp;
  int b = (int)(pix >> 12);
  int p = (int)(pix & 4095);
  int y = p >> 6, x = p & 63;
  int p0 = r / 2 - 1;
  int rr = r * r;
  float acc = bc[co];
  for (int dy = 0; dy < r; ++dy) {
    int sy = y + dy - p0;
    if (sy < 0 || sy >= 64) continue;
    for (int dx = 0; dx < r; ++dx) {
      int sx = x + dx - p0;
      if (sx < 0 || sx >= 64) continue;
      const float* xr = X + ((long long)b * 4096 + sy * 64 + sx) * 512;
      const float* wr = Wc + (long long)co * 512 * rr + dy * r + dx;
      float a = 0.f;
      for (int ci = 0; ci < 512; ++ci) a += xr[ci] * wr[(long long)ci * rr];
      acc += a;
    }
  }
  Y[pix * cp + co] = fmaxf(acc, 0.f);
}

// window unfold with wrap pad + space-to-depth -> Wn [32768 x 512]
__global__ void win_gather(const float* __restrict__ CTX, float* __restrict__ Wn,
                           int r, int cp)
{
  long long idx = (long long)blockIdx.x * 256 + threadIdx.x;
  if (idx >= 32768LL * 512) return;
  int f = (int)(idx % 512);
  long long row = idx / 512;
  int c = f % cp;
  int rcomb = f / cp;
  int rh = rcomb / r, rw = rcomb % r;
  int t = (int)(row & 63);
  long long w = row >> 6;
  int j = (int)(w & 7);
  int i = (int)((w >> 3) & 7);
  int b = (int)(w >> 6);
  int py = t >> 3, px = t & 7;
  int m  = 4 * (r - 1);
  int wl = 4 * (r + 1);
  int wrr = 8 * r;
  int er = i * 8 + py * r + rh;
  int ec = j * 8 + px * r + rw;
  int sy = (er < m) ? (wl + er) : ((er < m + 64) ? (er - m) : (64 - wrr + (er - m - 64)));
  int sx = (ec < m) ? (wl + ec) : ((ec < m + 64) ? (ec - m) : (64 - wrr + (ec - m - 64)));
  Wn[idx] = CTX[((long long)b * 4096 + sy * 64 + sx) * cp + c];
}

// bilinear 64x64 -> 128x128 (half-pixel centers, edge clamp), NHWC 512ch
__global__ void bilinear_up(const float* __restrict__ F, float* __restrict__ U)
{
  long long idx = (long long)blockIdx.x * 256 + threadIdx.x;
  if (idx >= 131072LL * 512) return;
  int c = (int)(idx % 512);
  long long pix = idx / 512;
  int b = (int)(pix >> 14);
  int p = (int)(pix & 16383);
  int Y = p >> 7, X = p & 127;
  float sy = fminf(fmaxf(Y * 0.5f - 0.25f, 0.f), 63.f);
  float sx = fminf(fmaxf(X * 0.5f - 0.25f, 0.f), 63.f);
  int y0 = (int)sy, x0 = (int)sx;
  int y1 = y0 < 63 ? y0 + 1 : 63;
  int x1 = x0 < 63 ? x0 + 1 : 63;
  float fy = sy - y0, fx = sx - x0;
  long long base = (long long)b * 4096;
  float v00 = F[(base + y0 * 64 + x0) * 512 + c];
  float v01 = F[(base + y0 * 64 + x1) * 512 + c];
  float v10 = F[(base + y1 * 64 + x0) * 512 + c];
  float v11 = F[(base + y1 * 64 + x1) * 512 + c];
  U[idx] = (1.f - fy) * ((1.f - fx) * v00 + fx * v01) +
           fy * ((1.f - fx) * v10 + fx * v11);
}

// --------------------------------- launcher --------------------------------

static void gemm(hipStream_t s, const float* A, int lda, const float* Bw, int ldb,
                 const float* bias, float* C, long long ldc, long long colOff,
                 int M, int N, int K, int transB, int relu, int accum,
                 int outNCHW, int HW)
{
  dim3 g(CDIV(N, TN), CDIV(M, TM));
  gemm_wmma<<<g, dim3(256), 0, s>>>(A, lda, Bw, ldb, bias, C, ldc, colOff,
                                    M, N, K, transB, relu, accum, outNCHW, HW);
}

extern "C" void kernel_launch(void* const* d_in, const int* in_sizes, int n_in,
                              void* d_out, int out_size, void* d_ws, size_t ws_size,
                              hipStream_t stream)
{
  (void)in_sizes; (void)n_in; (void)out_size; (void)ws_size;
  const float* c1   = (const float*)d_in[0];
  const float* c2   = (const float*)d_in[1];
  const float* c3   = (const float*)d_in[2];
  const float* c4   = (const float*)d_in[3];
  const float* W4   = (const float*)d_in[4];  const float* b4   = (const float*)d_in[5];
  const float* W3   = (const float*)d_in[6];  const float* b3   = (const float*)d_in[7];
  const float* W2   = (const float*)d_in[8];  const float* b2   = (const float*)d_in[9];
  const float* W1   = (const float*)d_in[10]; const float* b1   = (const float*)d_in[11];
  const float* Wf   = (const float*)d_in[12]; const float* bf   = (const float*)d_in[13];
  const float* Wsp  = (const float*)d_in[14]; const float* bsp  = (const float*)d_in[15];
  const float* Wip  = (const float*)d_in[16]; const float* bip  = (const float*)d_in[17];
  const float* Wcat = (const float*)d_in[18]; const float* bcat = (const float*)d_in[19];
  const float* Wlow = (const float*)d_in[20]; const float* blow = (const float*)d_in[21];

  float* ws = (float*)d_ws;
  size_t off = 0;
  auto alloc = [&](long long n) { float* p = ws + off; off += (size_t)n; return p; };

  float* T4   = alloc(2048LL * 768);
  float* T3   = alloc(8192LL * 384);
  float* T2   = alloc(32768LL * 192);
  float* A4   = alloc(2048LL * 512);
  float* A3   = alloc(8192LL * 512);
  float* ACAT = alloc(32768LL * 1536);   // later reused as QP/KP/VP
  float* CB   = alloc(32768LL * 512);    // _c
  float* CATB = alloc(32768LL * 2560);   // later reused as bilinear output
  float* MB   = alloc(8 * 512);
  float* PB   = alloc(8 * 512);
  float* QB   = alloc(32768LL * 512);    // later reused as _c1
  float* CTX  = alloc(32768LL * 128);
  float* WB   = alloc(32768LL * 512);
  float* OB   = alloc(32768LL * 512);
  float* OB2  = alloc(32768LL * 512);    // later reused as c1-NHWC
  float* FB   = alloc(32768LL * 512);

  float* QP  = ACAT;
  float* KPb = ACAT + 32768LL * 512;
  float* VPb = ACAT + 2 * 32768LL * 512;
  float* UB  = CATB;                     // 131072x512 fits in 32768x2560
  float* C1B = QB;                       // 131072x48  fits in 32768x512
  float* T1  = OB2;                      // 131072x96  fits in 32768x512
  float* OUT = (float*)d_out;

  tdm_probe<<<1, 32, 0, stream>>>();

  // ---- FPN: per-level conv1x1 + nearest upsample into concat buffer ----
  nchw_to_nhwc<<<CDIV(2048LL * 768, 256), 256, 0, stream>>>(c4, T4, 8, 768, 256);
  nchw_to_nhwc<<<CDIV(8192LL * 384, 256), 256, 0, stream>>>(c3, T3, 8, 384, 1024);
  nchw_to_nhwc<<<CDIV(32768LL * 192, 256), 256, 0, stream>>>(c2, T2, 8, 192, 4096);

  gemm(stream, T4, 768, W4, 512, b4, A4, 512, 0, 2048, 512, 768, 0, 1, 0, 0, 0);
  gemm(stream, T3, 384, W3, 512, b3, A3, 512, 0, 8192, 512, 384, 0, 1, 0, 0, 0);
  gemm(stream, T2, 192, W2, 512, b2, ACAT, 1536, 1024, 32768, 512, 192, 0, 1, 0, 0, 0);
  expand_nearest<<<CDIV(32768LL * 512, 256), 256, 0, stream>>>(A4, ACAT, 16, 4, 0);
  expand_nearest<<<CDIV(32768LL * 512, 256), 256, 0, stream>>>(A3, ACAT, 32, 2, 512);

  // _c = relu(concat @ Wf + bf)
  gemm(stream, ACAT, 1536, Wf, 512, bf, CB, 512, 0, 32768, 512, 1536, 0, 1, 0, 0, 0);

  // branch 0: spatial conv1x1 -> CAT[:, 0:512]
  gemm(stream, CB, 512, Wsp, 512, bsp, CATB, 2560, 0, 32768, 512, 512, 0, 1, 0, 0, 0);
  // branch 1: global pool -> CAT[:, 512:1024]
  mean_pix<<<CDIV(8 * 512, 256), 256, 0, stream>>>(CB, MB);
  gemm(stream, MB, 512, Wip, 512, bip, PB, 512, 0, 8, 512, 512, 0, 1, 0, 0, 0);
  bcast_pool<<<CDIV(32768LL * 512, 256), 256, 0, stream>>>(PB, CATB);

  // query windows (shared by all 3 attention levels)
  q_gather<<<CDIV(32768LL * 512, 256), 256, 0, stream>>>(CB, QB);

  // ---- 3 Lawin attention levels (r = 2, 4, 8) ----
  for (int j = 0; j < 3; ++j) {
    int r  = 2 << j;
    int cp = 512 / (r * r);
    const float* dsw = (const float*)d_in[22 + 6 * j];
    const float* dsb = (const float*)d_in[23 + 6 * j];
    const float* Wi  = (const float*)d_in[24 + 6 * j];
    const float* bi  = (const float*)d_in[25 + 6 * j];
    const float* Wo  = (const float*)d_in[26 + 6 * j];
    const float* bo  = (const float*)d_in[27 + 6 * j];

    conv_ds<<<CDIV(32768LL * cp, 256), 256, 0, stream>>>(CB, dsw, dsb, CTX, r, cp);
    win_gather<<<CDIV(32768LL * 512, 256), 256, 0, stream>>>(CTX, WB, r, cp);

    // Q/K/V projections: x @ Wi[slice].T + bi[slice]
    gemm(stream, QB, 512, Wi,              512, bi,        QP,  512, 0, 32768, 512, 512, 1, 0, 0, 0, 0);
    gemm(stream, WB, 512, Wi + 512 * 512,  512, bi + 512,  KPb, 512, 0, 32768, 512, 512, 1, 0, 0, 0, 0);
    gemm(stream, WB, 512, Wi + 1024 * 512, 512, bi + 1024, VPb, 512, 0, 32768, 512, 512, 1, 0, 0, 0, 0);

    attn64<<<4096, 256, 0, stream>>>(QP, KPb, VPb, OB);

    // output projection + un-window + residual into CAT slice
    gemm(stream, OB, 512, Wo, 512, bo, OB2, 512, 0, 32768, 512, 512, 1, 0, 0, 0, 0);
    unwin_add<<<CDIV(32768LL * 512, 256), 256, 0, stream>>>(OB2, CB, CATB, 1024 + 512 * j);
  }

  // fuse: relu(CAT @ Wcat + bcat)
  gemm(stream, CATB, 2560, Wcat, 512, bcat, FB, 512, 0, 32768, 512, 2560, 0, 1, 0, 0, 0);

  // bilinear 64 -> 128 (UB aliases CATB, which is now dead)
  bilinear_up<<<CDIV(131072LL * 512, 256), 256, 0, stream>>>(FB, UB);

  // low-level branch: _c1 = relu(c1 @ W1 + b1)
  nchw_to_nhwc<<<CDIV(131072LL * 96, 256), 256, 0, stream>>>(c1, T1, 8, 96, 16384);
  gemm(stream, T1, 96, W1, 48, b1, C1B, 48, 0, 131072, 48, 96, 0, 1, 0, 0, 0);

  // final fuse 560 -> 256, written NCHW to d_out, as two K-slice GEMMs
  gemm(stream, UB, 512, Wlow, 256, nullptr, OUT, 0, 0, 131072, 256, 512, 0, 0, 0, 1, 16384);
  gemm(stream, C1B, 48, Wlow + 512 * 256, 256, blow, OUT, 0, 0, 131072, 256, 48, 0, 1, 1, 1, 16384);
}
